// Mamba2Block_40398462386638
// MI455X (gfx1250) — matrix operand order repaired and checked by reading, not compiled, not run on hardware
//
#include <hip/hip_runtime.h>
#include <hip/hip_bf16.h>

// ---------------------------------------------------------------------------
// Mamba2 block forward for MI455X (gfx1250), wave32, WMMA bf16 path.
// D_MODEL=2048 D_STATE=64 D_CONV=4 N_HEADS=64 HEAD_DIM=64 CHUNK=256
// D_INNER=4096 XBC_DIM=4224 D_IN_PROJ=8384 BL=4096 (b=2, L=2048)
// GEMMs: Tensor-Data-Mover (TDM) staged, double-buffered LDS, bf16 WMMA.
// ---------------------------------------------------------------------------

typedef __attribute__((ext_vector_type(16))) __bf16 v16bf;
typedef __attribute__((ext_vector_type(8)))  __bf16 v8bf;
typedef __attribute__((ext_vector_type(8)))  float  v8f;
typedef __attribute__((ext_vector_type(4)))  unsigned u32x4;
typedef __attribute__((ext_vector_type(8)))  int     i32x8;
typedef __attribute__((ext_vector_type(4)))  int     i32x4;

#define BL       4096
#define DM       2048
#define DI       4096
#define NP       8384    // D_IN_PROJ
#define XBC      4224
#define NH       64
#define HD       64
#define DS       64
#define NCHUNKS  32      // 2 batches * 16 chunks
#define LSEQ     2048

#if defined(__has_builtin)
#if __has_builtin(__builtin_amdgcn_tensor_load_to_lds)
#define HAVE_TDM 1
#endif
#endif
#ifndef HAVE_TDM
#define HAVE_TDM 0
#endif

__device__ __forceinline__ __bf16 f2bf(float f) {
  union { float f; unsigned u; } v; v.f = f;
  unsigned r = (v.u + 0x7FFFu + ((v.u >> 16) & 1u)) >> 16;
  union { unsigned short s; __bf16 b; } o; o.s = (unsigned short)r;
  return o.b;
}
__device__ __forceinline__ float bf2f(__bf16 b) {
  union { __bf16 b; unsigned short s; } u; u.b = b;
  union { unsigned u; float f; } o; o.u = ((unsigned)u.s) << 16;
  return o.f;
}
__device__ __forceinline__ v8f vzero() {
  v8f z;
#pragma unroll
  for (int i = 0; i < 8; i++) z[i] = 0.0f;
  return z;
}
__device__ __forceinline__ v16bf mk_afrag(const __bf16* lo, const __bf16* hi) {
  v8bf a = *(const v8bf*)lo;
  v8bf b = *(const v8bf*)hi;
  return __builtin_shufflevector(a, b, 0,1,2,3,4,5,6,7,8,9,10,11,12,13,14,15);
}
__device__ __forceinline__ v8f wmma_bf16(v16bf a, v16bf b, v8f c) {
  return __builtin_amdgcn_wmma_f32_16x16x32_bf16(false, a, false, b, (short)0, c, false, false);
}

#if HAVE_TDM
// Issue one TDM 2-D tile load: tile_d1 rows x tile_d0 elems of bf16 (2B),
// row stride = stride0 elements, global tile start gptr, dest LDS byte addr.
// D# packing per CDNA5 ISA ch.8 (group0: count/lds/global/type, group1: dims).
__device__ __forceinline__ void tdm_load_2d(unsigned lds_off, const void* gptr,
                                            unsigned tensor_d0, unsigned tensor_d1,
                                            unsigned tile_d0, unsigned tile_d1,
                                            unsigned long long stride0) {
  unsigned long long ga = (unsigned long long)(uintptr_t)gptr;
  u32x4 g0;
  g0[0] = 1u;                                            // count=1 (valid user D#)
  g0[1] = lds_off;                                       // lds_addr (bytes)
  g0[2] = (unsigned)(ga & 0xFFFFFFFFull);                // global_addr[31:0]
  g0[3] = (unsigned)((ga >> 32) & 0x01FFFFFFull)         // global_addr[56:32]
          | (2u << 30);                                  // type = 2 ("image")
  i32x8 g1;
  g1[0] = (int)(1u << 16);                               // data_size=1 -> 2 bytes
  g1[1] = (int)((tensor_d0 & 0xFFFFu) << 16);            // tensor_dim0[15:0]
  g1[2] = (int)((tensor_d0 >> 16) | ((tensor_d1 & 0xFFFFu) << 16));
  g1[3] = (int)((tensor_d1 >> 16) | (tile_d0 << 16));    // tile_dim0
  g1[4] = (int)(tile_d1 & 0xFFFFu);                      // tile_dim1 (tile_dim2=0)
  g1[5] = (int)(stride0 & 0xFFFFFFFFull);                // tensor_dim0_stride lo
  g1[6] = (int)((stride0 >> 32) & 0xFFFFull);            // stride hi (dim1_stride=0)
  g1[7] = 0;
  i32x4 z4; z4[0] = 0; z4[1] = 0; z4[2] = 0; z4[3] = 0;
#if __clang_major__ >= 23
  i32x8 z8;
#pragma unroll
  for (int i = 0; i < 8; i++) z8[i] = 0;
  __builtin_amdgcn_tensor_load_to_lds(g0, g1, z4, z4, z8, 0);
#else
  __builtin_amdgcn_tensor_load_to_lds(g0, g1, z4, z4, 0);
#endif
}
#endif  // HAVE_TDM

// ------------------------------- fp32 -> bf16 -------------------------------
__global__ void k_cvt(const float* __restrict__ in, __bf16* __restrict__ out, int n) {
  int i = blockIdx.x * 256 + threadIdx.x;
  if (i < n) out[i] = f2bf(in[i]);
}

// ------------------------------- WMMA GEMM ----------------------------------
// C[M,N] (f32) = A[M,K] (bf16 row-major) * W[N,K]^T (bf16 row-major by N)
// Block: 128(M) x 256(N), 256 threads = 8 waves in 2x4, each wave 64x64.
// Tiles staged by TDM into double-buffered LDS; wave 0 drives the DMA.
__global__ void __launch_bounds__(256) k_gemm(const __bf16* __restrict__ A,
                                              const __bf16* __restrict__ W,
                                              float* __restrict__ C,
                                              int M, int N, int K) {
  __shared__ __bf16 As[2][128 * 32];
  __shared__ __bf16 Bs[2][256 * 32];
  const int t = threadIdx.x;
  const int lane = t & 31, wave = t >> 5;
  const int wm = wave >> 2, wn = wave & 3;
  const int bM = blockIdx.y * 128, bN = blockIdx.x * 256;
  const int nlo = lane & 15, khalf = lane >> 4;
  v8f acc[4][4];
#pragma unroll
  for (int i = 0; i < 4; i++)
#pragma unroll
    for (int j = 0; j < 4; j++) acc[i][j] = vzero();

#if HAVE_TDM
  const unsigned asoff[2] = {(unsigned)(uintptr_t)&As[0][0], (unsigned)(uintptr_t)&As[1][0]};
  const unsigned bsoff[2] = {(unsigned)(uintptr_t)&Bs[0][0], (unsigned)(uintptr_t)&Bs[1][0]};
  if (wave == 0) {  // prologue: DMA first K-slice into buffer 0
    tdm_load_2d(asoff[0], A + (size_t)bM * K, K, M - bM, 32, 128, (unsigned long long)K);
    tdm_load_2d(bsoff[0], W + (size_t)bN * K, K, N - bN, 32, 256, (unsigned long long)K);
  }
  int cur = 0;
  for (int k0 = 0; k0 < K; k0 += 32) {
    if (wave == 0) {
      if (k0 + 32 < K) {  // DMA next K-slice into the other buffer
        tdm_load_2d(asoff[cur ^ 1], A + (size_t)bM * K + (k0 + 32),
                    K - (k0 + 32), M - bM, 32, 128, (unsigned long long)K);
        tdm_load_2d(bsoff[cur ^ 1], W + (size_t)bN * K + (k0 + 32),
                    K - (k0 + 32), N - bN, 32, 256, (unsigned long long)K);
        __builtin_amdgcn_s_wait_tensorcnt(2);  // oldest 2 (current buffer) done
      } else {
        __builtin_amdgcn_s_wait_tensorcnt(0);
      }
    }
    __syncthreads();
    const __bf16* Asb = As[cur];
    const __bf16* Bsb = Bs[cur];
    v16bf af[4], bfv[4];
#pragma unroll
    for (int mt = 0; mt < 4; mt++) {
      const __bf16* p = Asb + (wm * 64 + mt * 16 + nlo) * 32 + khalf * 16;
      af[mt] = mk_afrag(p, p + 8);
    }
#pragma unroll
    for (int nt = 0; nt < 4; nt++)
      bfv[nt] = *(const v16bf*)(Bsb + (wn * 64 + nt * 16 + nlo) * 32 + khalf * 16);
#pragma unroll
    for (int mt = 0; mt < 4; mt++)
#pragma unroll
      for (int nt = 0; nt < 4; nt++)
        acc[mt][nt] = wmma_bf16(af[mt], bfv[nt], acc[mt][nt]);
    __syncthreads();  // all waves done reading before next DMA overwrites
    cur ^= 1;
  }
#else
  for (int k0 = 0; k0 < K; k0 += 32) {
    {  // A tile 128x32: thread -> half-row (16 bf16 = 2x uint4)
      int row = t >> 1, half = t & 1;
      const uint4* src = (const uint4*)(A + (size_t)(bM + row) * K + k0 + half * 16);
      uint4* dst = (uint4*)(&As[0][0] + row * 32 + half * 16);
      dst[0] = src[0];
      dst[1] = src[1];
    }
    {  // B tile 256x32: thread -> full row (32 bf16 = 4x uint4)
      int n = bN + t;
      int nc = n < N ? n : 0;
      const uint4* src = (const uint4*)(W + (size_t)nc * K + k0);
      uint4* dst = (uint4*)(&Bs[0][0] + t * 32);
      dst[0] = src[0]; dst[1] = src[1]; dst[2] = src[2]; dst[3] = src[3];
    }
    __syncthreads();
    v16bf af[4], bfv[4];
#pragma unroll
    for (int mt = 0; mt < 4; mt++) {
      const __bf16* p = &As[0][0] + (wm * 64 + mt * 16 + nlo) * 32 + khalf * 16;
      af[mt] = mk_afrag(p, p + 8);
    }
#pragma unroll
    for (int nt = 0; nt < 4; nt++)
      bfv[nt] = *(const v16bf*)(&Bs[0][0] + (wn * 64 + nt * 16 + nlo) * 32 + khalf * 16);
#pragma unroll
    for (int mt = 0; mt < 4; mt++)
#pragma unroll
      for (int nt = 0; nt < 4; nt++)
        acc[mt][nt] = wmma_bf16(af[mt], bfv[nt], acc[mt][nt]);
    __syncthreads();
  }
#endif
  const int rofs = khalf << 3;
#pragma unroll
  for (int mt = 0; mt < 4; mt++)
#pragma unroll
    for (int nt = 0; nt < 4; nt++)
#pragma unroll
      for (int r = 0; r < 8; r++) {
        int row = bM + wm * 64 + mt * 16 + r + rofs;
        int col = bN + wn * 64 + nt * 16 + nlo;
        if (col < N) C[(size_t)row * N + col] = acc[mt][nt][r];
      }
}

// ------------------------------- dt / A*dt ----------------------------------
__global__ void k_dt(const float* __restrict__ zx, const float* __restrict__ dt_bias,
                     const float* __restrict__ A_log,
                     float* __restrict__ dtv, float* __restrict__ ad) {
  int i = blockIdx.x * 256 + threadIdx.x;  // 0 .. BL*NH-1
  int r = i >> 6, h = i & 63;
  float x = zx[(size_t)r * NP + (2 * DI + 2 * DS) + h] + dt_bias[h];
  float sp = x > 20.0f ? x : log1pf(__expf(x));
  dtv[i] = sp;
  ad[i] = -__expf(A_log[h]) * sp;
}

// --------------------------- depthwise conv + silu ---------------------------
__global__ void __launch_bounds__(256) k_conv(const float* __restrict__ zx,
                                              const float* __restrict__ cw,
                                              const float* __restrict__ cb,
                                              const float* __restrict__ dtv,
                                              __bf16* __restrict__ xs,
                                              __bf16* __restrict__ Xd,
                                              __bf16* __restrict__ Bmo,
                                              __bf16* __restrict__ Cmo) {
  int r = blockIdx.x;
  int c = blockIdx.y * 256 + threadIdx.x;
  if (c >= XBC) return;
  int l = r & (LSEQ - 1);
  float acc = cb[c];
#pragma unroll
  for (int j = 0; j < 4; j++) {
    int lj = l - 3 + j;
    if (lj >= 0) acc += zx[(size_t)(r - 3 + j) * NP + DI + c] * cw[c * 4 + j];
  }
  float sv = acc / (1.0f + __expf(-acc));  // silu
  if (c < DI) {
    int h = c >> 6;
    xs[(size_t)r * DI + c] = f2bf(sv);
    Xd[(size_t)r * DI + c] = f2bf(sv * dtv[r * NH + h]);
  } else if (c < DI + DS) {
    Bmo[(size_t)r * DS + (c - DI)] = f2bf(sv);
  } else {
    Cmo[(size_t)r * DS + (c - DI - DS)] = f2bf(sv);
  }
}

// ------------------------- SSD intra-chunk (Y_diag + states) -----------------
// grid: (NCHUNKS, NH), 256 threads (8 waves).
__global__ void __launch_bounds__(256) k_ssd(const __bf16* __restrict__ Xd,
                                             const __bf16* __restrict__ Bm,
                                             const __bf16* __restrict__ Cm,
                                             const float* __restrict__ ad,
                                             float* __restrict__ Y,
                                             float* __restrict__ S,
                                             float* __restrict__ AcsG) {
  __shared__ float  Acs[256];
  __shared__ __bf16 Xt[64][256];      // X^T (p, l), X already * dt
  __shared__ __bf16 Btd_s[64 * 32];   // (B * decay)^T slice (n, 32 l's)
  __shared__ __bf16 Gs[8][16 * 32];   // per-wave masked-G scratch (A-fragment layout)
  const int t = threadIdx.x, lane = t & 31, wave = t >> 5;
  const int bc = blockIdx.x, h = blockIdx.y;
  const int rowbase = ((bc >> 4) * LSEQ) + (bc & 15) * 256;
  const int nlo = lane & 15, khalf = lane >> 4, rofs = khalf << 3;

  // ---- inclusive prefix scan of A*dt over the chunk ----
  float v = ad[(size_t)(rowbase + t) * NH + h];
  Acs[t] = v;
  __syncthreads();
  for (int ofs = 1; ofs < 256; ofs <<= 1) {
    float o = (t >= ofs) ? Acs[t - ofs] : 0.0f;
    __syncthreads();
    Acs[t] += o;
    __syncthreads();
  }
  AcsG[((size_t)bc * NH + h) * 256 + t] = Acs[t];
  const float last = Acs[255];

  // ---- stage X^T into LDS (16B vector loads, scatter stores) ----
  {
    union U8 { uint4 q; __bf16 e[8]; };
    const uint4* xr = (const uint4*)(Xd + (size_t)(rowbase + t) * DI + h * HD);
#pragma unroll
    for (int g8 = 0; g8 < 8; g8++) {
      U8 u; u.q = xr[g8];
#pragma unroll
      for (int q = 0; q < 8; q++) Xt[g8 * 8 + q][t] = u.e[q];
    }
  }
  __syncthreads();

  // ---- Y_diag: each wave takes strips (wave) and (15-wave) -> uniform load ----
#pragma unroll
  for (int si = 0; si < 2; si++) {
    const int s = (si == 0) ? wave : (15 - wave);
    const int ibase = s * 16;
    v8f yacc[4];
#pragma unroll
    for (int p = 0; p < 4; p++) yacc[p] = vzero();
    const int jmax = s >> 1;  // triangular: only K-slices with any j <= i
    for (int j2 = 0; j2 <= jmax; j2++) {
      const int jbase = j2 * 32;
      v8f g[2]; g[0] = vzero(); g[1] = vzero();
#pragma unroll
      for (int ks = 0; ks < 2; ks++) {  // state K = 64 -> 2x32
        const int k0 = ks * 32;
        const __bf16* cp = Cm + (size_t)(rowbase + ibase + nlo) * DS + k0 + khalf * 16;
        v16bf aC = mk_afrag(cp, cp + 8);
#pragma unroll
        for (int jt = 0; jt < 2; jt++) {
          const __bf16* bp = Bm + (size_t)(rowbase + jbase + jt * 16 + nlo) * DS + k0 + khalf * 16;
          v16bf bB = *(const v16bf*)bp;
          g[jt] = wmma_bf16(aC, bB, g[jt]);
        }
      }
      // mask with L = exp(Acs[i]-Acs[j]) for i>=j, write bf16 into wave scratch
#pragma unroll
      for (int jt = 0; jt < 2; jt++) {
        const int j = jbase + jt * 16 + nlo;
        const float aj = Acs[j];
#pragma unroll
        for (int r = 0; r < 8; r++) {
          const int il = ibase + r + rofs;
          float val = (j <= il) ? g[jt][r] * __expf(Acs[il] - aj) : 0.0f;
          Gs[wave][(r + rofs) * 32 + jt * 16 + nlo] = f2bf(val);
        }
      }
      // consume: Y_strip += G'_slice (16x32) @ X_slice (32x64)
      {
        const __bf16* gp = &Gs[wave][nlo * 32 + khalf * 16];
        v16bf aG = mk_afrag(gp, gp + 8);
#pragma unroll
        for (int pt = 0; pt < 4; pt++) {
          v16bf bX = *(const v16bf*)&Xt[pt * 16 + nlo][jbase + khalf * 16];
          yacc[pt] = wmma_bf16(aG, bX, yacc[pt]);
        }
      }
    }
#pragma unroll
    for (int pt = 0; pt < 4; pt++)
#pragma unroll
      for (int r = 0; r < 8; r++) {
        int row = rowbase + ibase + r + rofs;
        int col = h * HD + pt * 16 + nlo;
        Y[(size_t)row * DI + col] = yacc[pt][r];
      }
  }

  // ---- states[p,n] = sum_l X[l,p]*decay[l]*B[l,n] ; 16 tiles over 8 waves ----
  const int tid0 = wave * 2;
  v8f sacc[2];
  sacc[0] = vzero(); sacc[1] = vzero();
  for (int kk = 0; kk < 8; kk++) {
    const int kb = kk * 32;
    __syncthreads();
    {  // stage (B*decay)^T slice: rows n=0..63, cols l=kb..kb+31
      int n = t >> 2, seg = t & 3;
#pragma unroll
      for (int q = 0; q < 8; q++) {
        int l = kb + seg * 8 + q;
        Btd_s[n * 32 + seg * 8 + q] =
            f2bf(bf2f(Bm[(size_t)(rowbase + l) * DS + n]) * __expf(last - Acs[l]));
      }
    }
    __syncthreads();
#pragma unroll
    for (int ti = 0; ti < 2; ti++) {
      const int tid = tid0 + ti;
      const int pt = tid >> 2, nt = tid & 3;
      const __bf16* ap = &Xt[pt * 16 + nlo][kb + khalf * 16];
      v16bf aX = mk_afrag(ap, ap + 8);
      v16bf bB = *(const v16bf*)&Btd_s[(nt * 16 + nlo) * 32 + khalf * 16];
      sacc[ti] = wmma_bf16(aX, bB, sacc[ti]);
    }
  }
#pragma unroll
  for (int ti = 0; ti < 2; ti++) {
    const int tid = tid0 + ti;
    const int pt = tid >> 2, nt = tid & 3;
#pragma unroll
    for (int r = 0; r < 8; r++) {
      int p = pt * 16 + r + rofs;
      int n = nt * 16 + nlo;
      S[(((size_t)bc * NH + h) * HD + p) * DS + n] = sacc[ti][r];
    }
  }
}

// ----------------------- inter-chunk state recurrence ------------------------
__global__ void __launch_bounds__(256) k_rec(const float* __restrict__ S,
                                             const float* __restrict__ AcsG,
                                             __bf16* __restrict__ Sin) {
  const int bh = blockIdx.x;  // b*64+h
  const int b = bh >> 6, h = bh & 63;
  const int t = threadIdx.x;
  float run[16];
#pragma unroll
  for (int j = 0; j < 16; j++) run[j] = 0.0f;
  for (int c = 0; c < 16; c++) {
    const int bc = b * 16 + c;
    const size_t base = (((size_t)bc * NH) + h) * (HD * DS);
    const float esum = __expf(AcsG[(((size_t)bc * NH) + h) * 256 + 255]);
#pragma unroll
    for (int j = 0; j < 16; j++) {
      const int e = t + j * 256;
      Sin[base + e] = f2bf(run[j]);
      run[j] = esum * run[j] + S[base + e];
    }
  }
}

// ----------------- Y_off = exp(Acs)* (C @ Sin^T)  +  xs*D  -------------------
__global__ void __launch_bounds__(256) k_yoff(const __bf16* __restrict__ Cm,
                                              const __bf16* __restrict__ Sin,
                                              const float* __restrict__ AcsG,
                                              const __bf16* __restrict__ xs,
                                              const float* __restrict__ Dp,
                                              float* __restrict__ Y) {
  const int t = threadIdx.x, lane = t & 31, wave = t >> 5;
  const int bc = blockIdx.x, h = blockIdx.y;
  const int rowbase = ((bc >> 4) * LSEQ) + (bc & 15) * 256;
  const int nlo = lane & 15, khalf = lane >> 4, rofs = khalf << 3;
  const float dv = Dp[h];
  const size_t sbase = (((size_t)bc * NH) + h) * (HD * DS);
  const float* acs = AcsG + (((size_t)bc * NH) + h) * 256;
#pragma unroll
  for (int si = 0; si < 2; si++) {
    const int ibase = (wave + si * 8) * 16;
    v8f acc[4];
#pragma unroll
    for (int p = 0; p < 4; p++) acc[p] = vzero();
#pragma unroll
    for (int ks = 0; ks < 2; ks++) {
      const int k0 = ks * 32;
      const __bf16* cp = Cm + (size_t)(rowbase + ibase + nlo) * DS + k0 + khalf * 16;
      v16bf aC = mk_afrag(cp, cp + 8);
#pragma unroll
      for (int pt = 0; pt < 4; pt++) {
        const __bf16* sp = Sin + sbase + (size_t)(pt * 16 + nlo) * DS + k0 + khalf * 16;
        v16bf bS = *(const v16bf*)sp;
        acc[pt] = wmma_bf16(aC, bS, acc[pt]);
      }
    }
#pragma unroll
    for (int pt = 0; pt < 4; pt++)
#pragma unroll
      for (int r = 0; r < 8; r++) {
        const int i = ibase + r + rofs;
        const int row = rowbase + i;
        const int col = h * HD + pt * 16 + nlo;
        const size_t idx = (size_t)row * DI + col;
        Y[idx] += acc[pt][r] * __expf(acs[i]) + bf2f(xs[idx]) * dv;
      }
  }
}

// ---------------------- gating (z*silu(y)) + RMSNorm -------------------------
__global__ void __launch_bounds__(256) k_gate(const float* __restrict__ zx,
                                              const float* __restrict__ Y,
                                              const float* __restrict__ nw,
                                              __bf16* __restrict__ yn) {
  __shared__ float red[256];
  const int row = blockIdx.x, t = threadIdx.x;
  float loc[16];
  float s = 0.0f;
#pragma unroll
  for (int j = 0; j < 16; j++) {
    const int col = t + j * 256;
    const float y = Y[(size_t)row * DI + col];
    const float z = zx[(size_t)row * NP + col];
    const float g = z * y / (1.0f + __expf(-y));
    loc[j] = g;
    s += g * g;
  }
  red[t] = s;
  __syncthreads();
  for (int o = 128; o > 0; o >>= 1) {
    if (t < o) red[t] += red[t + o];
    __syncthreads();
  }
  const float rms = rsqrtf(red[0] * (1.0f / (float)DI) + 1.1920929e-07f);
#pragma unroll
  for (int j = 0; j < 16; j++) {
    const int col = t + j * 256;
    yn[(size_t)row * DI + col] = f2bf(loc[j] * rms * nw[col]);
  }
}

// ---------------------------------------------------------------------------
extern "C" void kernel_launch(void* const* d_in, const int* in_sizes, int n_in,
                              void* d_out, int out_size, void* d_ws, size_t ws_size,
                              hipStream_t stream) {
  (void)in_sizes; (void)n_in; (void)out_size; (void)ws_size;
  const float* x       = (const float*)d_in[0];
  const float* w1      = (const float*)d_in[1];
  const float* cw      = (const float*)d_in[2];
  const float* cb      = (const float*)d_in[3];
  const float* dt_bias = (const float*)d_in[4];
  const float* A_log   = (const float*)d_in[5];
  const float* Dp      = (const float*)d_in[6];
  const float* nw      = (const float*)d_in[7];
  const float* w2      = (const float*)d_in[8];
  float* out = (float*)d_out;

  char* ws = (char*)d_ws;
  size_t off = 0;
  auto alloc = [&](size_t bytes) -> void* {
    void* p = ws + off;
    off += (bytes + 255) & ~(size_t)255;
    return p;
  };
  float*  zx    = (float*)alloc((size_t)BL * NP * 4);           // in_proj output
  __bf16* xbf   = (__bf16*)alloc((size_t)BL * DM * 2);
  __bf16* w1bf  = (__bf16*)alloc((size_t)NP * DM * 2);
  __bf16* w2bf  = (__bf16*)alloc((size_t)DM * DI * 2);
  float*  dtv   = (float*)alloc((size_t)BL * NH * 4);
  float*  ad    = (float*)alloc((size_t)BL * NH * 4);
  __bf16* xs    = (__bf16*)alloc((size_t)BL * DI * 2);
  __bf16* Xd    = (__bf16*)alloc((size_t)BL * DI * 2);
  __bf16* Bm    = (__bf16*)alloc((size_t)BL * DS * 2);
  __bf16* Cm    = (__bf16*)alloc((size_t)BL * DS * 2);
  float*  AcsG  = (float*)alloc((size_t)NCHUNKS * NH * 256 * 4);
  float*  Yb    = (float*)alloc((size_t)BL * DI * 4);
  float*  S     = (float*)alloc((size_t)NCHUNKS * NH * HD * DS * 4);
  __bf16* Sin   = (__bf16*)alloc((size_t)NCHUNKS * NH * HD * DS * 2);
  __bf16* yn    = (__bf16*)alloc((size_t)BL * DI * 2);

  // 1. convert activations / weights to bf16
  k_cvt<<<dim3((BL * DM + 255) / 256), dim3(256), 0, stream>>>(x, xbf, BL * DM);
  k_cvt<<<dim3((NP * DM + 255) / 256), dim3(256), 0, stream>>>(w1, w1bf, NP * DM);
  k_cvt<<<dim3((DM * DI + 255) / 256), dim3(256), 0, stream>>>(w2, w2bf, DM * DI);

  // 2. in_proj GEMM: [4096,2048] x [8384,2048]^T -> zx
  k_gemm<<<dim3((NP + 255) / 256, BL / 128), dim3(256), 0, stream>>>(xbf, w1bf, zx, BL, NP, DM);

  // 3. dt softplus and A*dt
  k_dt<<<dim3(BL * NH / 256), dim3(256), 0, stream>>>(zx, dt_bias, A_log, dtv, ad);

  // 4. depthwise conv + silu, emit xs, X*dt, B, C (bf16)
  k_conv<<<dim3(BL, (XBC + 255) / 256), dim3(256), 0, stream>>>(zx, cw, cb, dtv, xs, Xd, Bm, Cm);

  // 5. intra-chunk SSD (Y_diag + per-chunk states)
  k_ssd<<<dim3(NCHUNKS, NH), dim3(256), 0, stream>>>(Xd, Bm, Cm, ad, Yb, S, AcsG);

  // 6. inter-chunk recurrence -> states_in (bf16)
  k_rec<<<dim3(2 * NH), dim3(256), 0, stream>>>(S, AcsG, Sin);

  // 7. Y_off + skip connection
  k_yoff<<<dim3(NCHUNKS, NH), dim3(256), 0, stream>>>(Cm, Sin, AcsG, xs, Dp, Yb);

  // 8. gate + RMSNorm -> yn (bf16)
  k_gate<<<dim3(BL), dim3(256), 0, stream>>>(zx, Yb, nw, yn);

  // 9. out_proj GEMM: [4096,4096] x [2048,4096]^T -> out
  k_gemm<<<dim3(DM / 256, BL / 128), dim3(256), 0, stream>>>(yn, w2bf, out, BL, DM, DI);
}
